// MLPKAN_19129784336543
// MI455X (gfx1250) — matrix operand order repaired and checked by reading, not compiled, not run on hardware
//
#include <hip/hip_runtime.h>
#include <hip/hip_bf16.h>
#include <stdint.h>

#define BATCH 2048
#define DIN   128
#define DH    128

typedef __attribute__((ext_vector_type(16))) _Float16 v16h;
typedef __attribute__((ext_vector_type(8)))  float    v8f;
typedef __attribute__((ext_vector_type(4)))  unsigned int v4u;
typedef __attribute__((ext_vector_type(8)))  unsigned int v8u;

__device__ __forceinline__ float silu_f(float v) {
    // x * sigmoid(x); v_exp_f32 + v_rcp_f32 on the trans pipe
    return v / (1.0f + __expf(-v));
}

// One 1->5->5->1 MLP on scalar t. Weight pointers are block-uniform so the
// compiler emits scalar (SMEM) loads broadcast across the wave.
__device__ __forceinline__ float mlp_eval(const float* __restrict__ w0,
                                          const float* __restrict__ b0,
                                          const float* __restrict__ w1,
                                          const float* __restrict__ b1,
                                          const float* __restrict__ w2,
                                          float b2, float t) {
    float a[5];
#pragma unroll
    for (int k = 0; k < 5; ++k) a[k] = silu_f(fmaf(w0[k], t, b0[k]));
    float c[5];
#pragma unroll
    for (int u = 0; u < 5; ++u) {
        float acc = b1[u];
#pragma unroll
        for (int v = 0; v < 5; ++v) acc = fmaf(w1[u * 5 + v], a[v], acc);
        c[u] = silu_f(acc);
    }
    float z = b2;
#pragma unroll
    for (int v = 0; v < 5; ++v) z = fmaf(w2[v], c[v], z);
    return z;
}

// Kernel 1: layer-0 per-net MLPs.  partial[b][j] = sum_i mlp_{i,j}(x[b,i]) * s[i*128+j]
// Grid: (DH, BATCH/256). One thread = one batch element, j fixed per block
// -> net index n = i*DH+j is block-uniform -> weights via scalar loads.
// x staging: CDNA5 async memory->LDS (GLOBAL_LOAD_ASYNC_TO_LDS_B64, ASYNCcnt),
// row stride 34 floats: 8B-aligned rows, conflict-free strided reads.
__global__ __launch_bounds__(256) void kan_layer0_mlp(
    const float* __restrict__ x,
    const float* __restrict__ w0, const float* __restrict__ b0,
    const float* __restrict__ w1, const float* __restrict__ b1,
    const float* __restrict__ w2, const float* __restrict__ b2,
    const float* __restrict__ s,
    float* __restrict__ partial) {
    const int j   = blockIdx.x;           // output feature, uniform
    const int bb0 = blockIdx.y * 256;     // batch tile base
    const int bl  = threadIdx.x;          // local batch index

    __shared__ float xS[256 * 34];        // ~34 KB of the 320 KB WGP LDS

    float acc = 0.0f;
    for (int ic = 0; ic < DIN; ic += 32) {
        __syncthreads();                  // previous chunk fully consumed
#pragma unroll
        for (int p = 0; p < 16; ++p) {
            const int pkt = bl + p * 256;         // 0..4095
            const int bb  = pkt >> 4;             // tile row 0..255
            const int q   = pkt & 15;             // 8B packet within row
            const uint64_t gaddr =
                (uint64_t)(uintptr_t)(x + (size_t)(bb0 + bb) * DIN + ic + q * 2);
            // low 32 bits of a generic LDS pointer == LDS byte offset (ISA 10.2)
            const uint32_t laddr = (uint32_t)(uintptr_t)(&xS[bb * 34 + q * 2]);
            asm volatile("global_load_async_to_lds_b64 %0, %1, off"
                         :: "v"(laddr), "v"(gaddr) : "memory");
        }
        asm volatile("s_wait_asynccnt 0x0" ::: "memory"); // our packets landed
        __syncthreads();                  // everyone's packets visible

#pragma unroll 4
        for (int ii = 0; ii < 32; ++ii) {
            const int i = ic + ii;
            const int n = i * DH + j;     // block-uniform net index
            const float t = xS[bl * 34 + ii];
            const float z = mlp_eval(w0 + n * 5, b0 + n * 5,
                                     w1 + n * 25, b1 + n * 5,
                                     w2 + n * 5, b2[n], t);
            acc = fmaf(z, s[n], acc);
        }
    }
    partial[(bb0 + bl) * DH + j] = acc;
}

// Kernel 2: h = partial + x @ R  via v_wmma_f32_16x16x32_f16.
// One block = one 16-row batch tile x all 8 column tiles (8 waves).
// The whole 128x128 f32 R matrix (64 KB) is DMA'd into LDS once per block by
// the Tensor Data Mover (tensor_load_to_lds, TENSORcnt), replacing per-wave
// 512B-strided global column reads with unit-stride conflict-free LDS reads.
// D# descriptor per ISA 8.3/8.4 (2D tensor, groups 2/3 NULL):
//   g0 = { count=1, lds_addr, gaddr[31:0], gaddr[56:32] | type=2<<30 }
//   g1 = { data_size=4B, dim0=128, dim1=128, tile0=128, tile1=128, stride0=128 }
// WMMA layouts as before (A 16x32 f16 per ISA 7.12.2, B by symmetry, C/D f32).
__global__ __launch_bounds__(256) void kan_layer0_rgemm(
    const float* __restrict__ x,      // [BATCH, DIN]
    const float* __restrict__ r,      // R[i][j] = r0[i*DH + j]
    float* __restrict__ hbuf) {       // in: partial, out: h  [BATCH, DH]
    const int lane = threadIdx.x & 31;
    const int wave = threadIdx.x >> 5;
    const int tm = blockIdx.x;            // BATCH/16 = 128 row tiles
    const int tn = wave;                  // DH/16   = 8   col tiles
    const int half = lane >> 4, l16 = lane & 15;
    const int rowBase = tm * 16, colBase = tn * 16;

    __shared__ float rS[DIN * DH];        // 64 KB: full R matrix, row-major

    if (wave == 0) {                      // one TDM descriptor per block
        const uint64_t gaddr = (uint64_t)(uintptr_t)r;
        const uint32_t lds0  = (uint32_t)(uintptr_t)rS;
        const v4u g0 = { 1u,                                  // count=1, no gather
                         lds0,                                // lds_addr
                         (uint32_t)gaddr,                     // global_addr[31:0]
                         (uint32_t)(gaddr >> 32) | (2u << 30) // [56:32] | type=2
                       };
        const v8u g1 = { 0x00020000u,      // wg_mask=0, data_size=2 (4B)
                         128u << 16,       // tensor_dim0[15:0] in [31:16]
                         128u << 16,       // tensor_dim0 hi=0, tensor_dim1 lo=128
                         128u << 16,       // tensor_dim1 hi=0, tile_dim0=128
                         128u,             // tile_dim1=128, tile_dim2=0
                         128u,             // tensor_dim0_stride lo = 128
                         0u, 0u };         // stride hi / dim1_stride = 0
        asm volatile("tensor_load_to_lds %0, %1" :: "s"(g0), "s"(g1) : "memory");
        __builtin_amdgcn_s_wait_tensorcnt(0);   // s_wait_tensorcnt 0
    }
    __syncthreads();                      // R visible to all 8 waves

    v8f c;
#pragma unroll
    for (int v = 0; v < 8; ++v)
        c[v] = hbuf[(rowBase + v + half * 8) * DH + colBase + l16];

#pragma unroll
    for (int k = 0; k < DIN; k += 32) {
        v16h a, b;
        const float* xrow = x + (rowBase + l16) * DIN + k;
#pragma unroll
        for (int g = 0; g < 8; ++g) {
            const int kb = (g < 4) ? (2 * g + 8 * half)
                                   : (16 + 2 * (g - 4) + 8 * half);
            a[2 * g]     = (_Float16)xrow[kb];
            a[2 * g + 1] = (_Float16)xrow[kb + 1];
        }
#pragma unroll
        for (int g = 0; g < 8; ++g) {
            const int kk = k + ((g < 4) ? (2 * g + 8 * half)
                                        : (16 + 2 * (g - 4) + 8 * half));
            b[2 * g]     = (_Float16)rS[kk * DH + colBase + l16];
            b[2 * g + 1] = (_Float16)rS[(kk + 1) * DH + colBase + l16];
        }
        c = __builtin_amdgcn_wmma_f32_16x16x32_f16(
                /*neg_a=*/false, a, /*neg_b=*/false, b,
                /*c_mod=*/(short)0, c, /*reuse_a=*/false, /*reuse_b=*/false);
    }

#pragma unroll
    for (int v = 0; v < 8; ++v)
        hbuf[(rowBase + v + half * 8) * DH + colBase + l16] = c[v];
}

// Kernel 3: layer 1 (128 nets -> 1 output). Tiny (1% of FLOPs) -> plain VALU.
__global__ __launch_bounds__(256) void kan_layer1(
    const float* __restrict__ h,
    const float* __restrict__ w0, const float* __restrict__ b0,
    const float* __restrict__ w1, const float* __restrict__ b1,
    const float* __restrict__ w2, const float* __restrict__ b2,
    const float* __restrict__ s, const float* __restrict__ r,
    float* __restrict__ out) {
    const int b = blockIdx.x * 256 + threadIdx.x;
    float acc = 0.0f;
    for (int i = 0; i < DH; ++i) {        // n = i (out_size == 1), uniform
        const float t = h[b * DH + i];
        const float z = mlp_eval(w0 + i * 5, b0 + i * 5,
                                 w1 + i * 25, b1 + i * 5,
                                 w2 + i * 5, b2[i], t);
        acc = fmaf(z, s[i], fmaf(t, r[i], acc));
    }
    out[b] = acc;
}

extern "C" void kernel_launch(void* const* d_in, const int* in_sizes, int n_in,
                              void* d_out, int out_size, void* d_ws, size_t ws_size,
                              hipStream_t stream) {
    // setup_inputs() dict order:
    // 0:x 1:w0_0 2:b0_0 3:w0_1 4:b0_1 5:w0_2 6:b0_2 7:s0 8:r0
    // 9:w1_0 10:b1_0 11:w1_1 12:b1_1 13:w1_2 14:b1_2 15:s1 16:r1
    const float* x   = (const float*)d_in[0];
    const float* w00 = (const float*)d_in[1];
    const float* b00 = (const float*)d_in[2];
    const float* w01 = (const float*)d_in[3];
    const float* b01 = (const float*)d_in[4];
    const float* w02 = (const float*)d_in[5];
    const float* b02 = (const float*)d_in[6];
    const float* s0  = (const float*)d_in[7];
    const float* r0  = (const float*)d_in[8];
    const float* w10 = (const float*)d_in[9];
    const float* b10 = (const float*)d_in[10];
    const float* w11 = (const float*)d_in[11];
    const float* b11 = (const float*)d_in[12];
    const float* w12 = (const float*)d_in[13];
    const float* b12 = (const float*)d_in[14];
    const float* s1  = (const float*)d_in[15];
    const float* r1  = (const float*)d_in[16];

    float* hbuf = (float*)d_ws;           // BATCH*DH f32 = 1 MB scratch

    dim3 g1(DH, BATCH / 256);
    kan_layer0_mlp<<<g1, 256, 0, stream>>>(x, w00, b00, w01, b01, w02, b02, s0, hbuf);

    kan_layer0_rgemm<<<BATCH / 16, 256, 0, stream>>>(x, r0, hbuf);

    kan_layer1<<<BATCH / 256, 256, 0, stream>>>(hbuf, w10, b10, w11, b11, w12, b12,
                                                s1, r1, (float*)d_out);
}
